// StatementEncoder_11897059410447
// MI455X (gfx1250) — compile-verified, matrix operand order checked by reading, not run on hardware
//
#include <hip/hip_runtime.h>
#include <hip/hip_bf16.h>

// ---------------------------------------------------------------------------
// Problem constants (match reference)
// ---------------------------------------------------------------------------
#define NNODES 131072
#define NEDGES 131071
#define DFEAT  128
#define K0SEL  91751      // ceil(0.7*N)
#define K1SEL  45876      // ceil(0.5*K0)

typedef _Float16 v16h __attribute__((ext_vector_type(16)));
typedef _Float16 v8h  __attribute__((ext_vector_type(8)));
typedef _Float16 v4h  __attribute__((ext_vector_type(4)));
typedef float    v8f  __attribute__((ext_vector_type(8)));
typedef float    v4f  __attribute__((ext_vector_type(4)));

// ---------------------------------------------------------------------------
// Helpers: monotonic float<->uint key (for atomic max / radix select)
// ---------------------------------------------------------------------------
__device__ __forceinline__ unsigned f2key(float f) {
  unsigned u = __float_as_uint(f);
  return (u & 0x80000000u) ? ~u : (u | 0x80000000u);
}
__device__ __forceinline__ float key2f(unsigned k) {
  return __uint_as_float((k & 0x80000000u) ? (k & 0x7FFFFFFFu) : ~k);
}

// Edge decode: e < Emax -> real edge (valid if s!=d, or e < *ecount when
// ecount non-null i.e. pre-compacted list); e >= Emax -> self loop.
__device__ __forceinline__ bool edge_decode(long long e, const int* __restrict__ src,
                                            const int* __restrict__ dst,
                                            const int* ecount, int Emax,
                                            int& s, int& d) {
  if (e < Emax) {
    if (ecount) {
      if (e >= (long long)(*ecount)) return false;
      s = src[e]; d = dst[e]; return true;
    }
    s = src[e]; d = dst[e]; return s != d;
  }
  s = d = (int)(e - Emax);
  return true;
}

// ---------------------------------------------------------------------------
// WMMA GEMM:  C[M,Nc] = A[M,K] * B[K,Nc]   (fp32 in/out, f16 multiply)
// block = 256 threads (8 waves); block tile 64x64; wave -> two 16x16 tiles.
// Register double-buffered staging; all LDS traffic is b64/b128.
// ---------------------------------------------------------------------------
#define TM 64
#define TN 64
#define TK 32

struct StageRegs {
  v4f a[2];       // A: float4 along K
  float b[2][4];  // B: 4 coalesced scalars along K (for transposed LDS store)
};

// A tile: 64 rows x 32 k = 512 float4; thread handles 2 (idx = tid + c*256)
//   ar = idx>>3 (0..63), ak = (idx&7)*4
// B tile: 32 k x 64 n = 512 groups of 4k; bn = idx&63, bk4 = (idx>>6)*4
__device__ __forceinline__ void load_tile(StageRegs& rg, const float* __restrict__ A,
                                          const float* __restrict__ B, int M, int K,
                                          int Nc, int m0, int n0, int k0, int tid) {
#pragma unroll
  for (int c = 0; c < 2; ++c) {
    int idx = tid + c * 256;
    int ar = idx >> 3, ak = (idx & 7) * 4;
    int gr = m0 + ar;
    if (gr >= M) gr = M - 1;  // clamp: garbage rows never stored
    rg.a[c] = *(const v4f*)&A[(size_t)gr * K + (k0 + ak)];
    int bn = idx & 63, bk4 = (idx >> 6) * 4;
    const float* bp = &B[(size_t)(k0 + bk4) * Nc + (n0 + bn)];
#pragma unroll
    for (int j = 0; j < 4; ++j) rg.b[c][j] = bp[(size_t)j * Nc];
  }
}

__device__ __forceinline__ void store_tile(const StageRegs& rg,
                                           _Float16 (*As)[TK + 8],
                                           _Float16 (*Bs)[TK + 8], int tid) {
#pragma unroll
  for (int c = 0; c < 2; ++c) {
    int idx = tid + c * 256;
    int ar = idx >> 3, ak = (idx & 7) * 4;
    v4h ah;
#pragma unroll
    for (int j = 0; j < 4; ++j) ah[j] = (_Float16)rg.a[c][j];
    *(v4h*)&As[ar][ak] = ah;                       // ds_store_b64
    int bn = idx & 63, bk4 = (idx >> 6) * 4;
    v4h bh;
#pragma unroll
    for (int j = 0; j < 4; ++j) bh[j] = (_Float16)rg.b[c][j];
    *(v4h*)&Bs[bn][bk4] = bh;                      // ds_store_b64 (transposed)
  }
}

__global__ __launch_bounds__(256) void k_gemm_wmma(const float* __restrict__ A,
                                                   const float* __restrict__ B,
                                                   float* __restrict__ C,
                                                   int M, int K, int Nc) {
  __shared__ __align__(16) _Float16 As[TM][TK + 8];   // 40-half row stride
  __shared__ __align__(16) _Float16 Bs[TN][TK + 8];   // [n][k] (transposed)

  const int tid  = threadIdx.x;
  const int wave = tid >> 5;
  const int lane = tid & 31;
  const int m0   = blockIdx.x * TM;
  const int n0   = blockIdx.y * TN;
  const int trow  = wave >> 1;          // 0..3
  const int tcol2 = (wave & 1) * 2;     // 0 or 2

  v8f acc0 = {}; v8f acc1 = {};
  const int r     = lane & 15;
  const int kbase = (lane < 16) ? 0 : 8;

  StageRegs rg;
  load_tile(rg, A, B, M, K, Nc, m0, n0, 0, tid);

  for (int k0 = 0; k0 < K; k0 += TK) {
    store_tile(rg, As, Bs, tid);
    __syncthreads();

    // issue next tile's global loads before the math (latency overlap)
    if (k0 + TK < K) load_tile(rg, A, B, M, K, Nc, m0, n0, k0 + TK, tid);
    if (k0 + 2 * TK < K) {
      int idx = tid;
      int ar = idx >> 3, ak = (idx & 7) * 4;
      int gr = m0 + ar; if (gr >= M) gr = M - 1;
      __builtin_prefetch(&A[(size_t)gr * K + (k0 + 2 * TK + ak)], 0, 0);
      int bn = idx & 63, bk4 = (idx >> 6) * 4;
      __builtin_prefetch(&B[(size_t)(k0 + 2 * TK + bk4) * Nc + (n0 + bn)], 0, 0);
    }

    // fragments: lane r holds row/col = r, halves map K = (h<8?h:h+8)+kbase
    int arow = trow * 16 + r;
    v8h alo = *(const v8h*)&As[arow][kbase];
    v8h ahi = *(const v8h*)&As[arow][kbase + 16];
    v16h afrag = __builtin_shufflevector(alo, ahi, 0,1,2,3,4,5,6,7,8,9,10,11,12,13,14,15);

    int bcol0 = (tcol2 + 0) * 16 + r;
    int bcol1 = (tcol2 + 1) * 16 + r;
    v8h b0lo = *(const v8h*)&Bs[bcol0][kbase];
    v8h b0hi = *(const v8h*)&Bs[bcol0][kbase + 16];
    v16h bfrag0 = __builtin_shufflevector(b0lo, b0hi, 0,1,2,3,4,5,6,7,8,9,10,11,12,13,14,15);
    v8h b1lo = *(const v8h*)&Bs[bcol1][kbase];
    v8h b1hi = *(const v8h*)&Bs[bcol1][kbase + 16];
    v16h bfrag1 = __builtin_shufflevector(b1lo, b1hi, 0,1,2,3,4,5,6,7,8,9,10,11,12,13,14,15);

    acc0 = __builtin_amdgcn_wmma_f32_16x16x32_f16(false, afrag, false, bfrag0,
                                                  (short)0, acc0, false, false);
    acc1 = __builtin_amdgcn_wmma_f32_16x16x32_f16(false, afrag, false, bfrag1,
                                                  (short)0, acc1, false, false);
    __syncthreads();
  }

  // C layout: vgpr i -> row i + (lane<16?0:8), col = lane&15
  int ccol0 = n0 + tcol2 * 16 + r;
  int ccol1 = ccol0 + 16;
  int rbase = m0 + trow * 16 + ((lane < 16) ? 0 : 8);
#pragma unroll
  for (int i = 0; i < 8; ++i) {
    int gr = rbase + i;
    if (gr < M) {
      C[(size_t)gr * Nc + ccol0] = acc0[i];
      C[(size_t)gr * Nc + ccol1] = acc1[i];
    }
  }
}

// ---------------------------------------------------------------------------
// Attention coefficients: one wave per (node, head)
// ---------------------------------------------------------------------------
__global__ __launch_bounds__(256) void k_att(const float* __restrict__ xh,
                                             const float* __restrict__ asrc,
                                             const float* __restrict__ adst,
                                             float* __restrict__ al_s,
                                             float* __restrict__ al_d,
                                             int n, int H) {
  long long gid = (long long)blockIdx.x * blockDim.x + threadIdx.x;
  long long wv  = gid >> 5;
  int lane = threadIdx.x & 31;
  if (wv >= (long long)n * H) return;
  int node = (int)(wv / H), h = (int)(wv % H);
  const float* row = xh + ((size_t)node * H + h) * DFEAT;
  float ss = 0.f, sd = 0.f;
  for (int c = lane; c < DFEAT; c += 32) {
    float v = row[c];
    ss += v * asrc[h * DFEAT + c];
    sd += v * adst[h * DFEAT + c];
  }
  for (int off = 16; off; off >>= 1) {
    ss += __shfl_down(ss, off);
    sd += __shfl_down(sd, off);
  }
  if (lane == 0) {
    al_s[(size_t)node * H + h] = ss;
    al_d[(size_t)node * H + h] = sd;
  }
}

// ---------------------------------------------------------------------------
// Edge logits + segment max (via uint-key atomic max)
// ---------------------------------------------------------------------------
__global__ void k_logits(const float* __restrict__ al_s, const float* __restrict__ al_d,
                         const int* __restrict__ src, const int* __restrict__ dst,
                         const int* ecount, int Emax, int n, int H,
                         float* __restrict__ logit, unsigned* __restrict__ mxkey) {
  long long t = (long long)blockIdx.x * blockDim.x + threadIdx.x;
  long long total = (long long)(Emax + n) * H;
  if (t >= total) return;
  int h = (int)(t % H);
  long long e = t / H;
  int s, d;
  float lg = -1e30f;
  if (edge_decode(e, src, dst, ecount, Emax, s, d)) {
    float x = al_s[(size_t)s * H + h] + al_d[(size_t)d * H + h];
    lg = (x > 0.f) ? x : 0.2f * x;               // leaky_relu(0.2)
    atomicMax(&mxkey[(size_t)d * H + h], f2key(lg));
  }
  logit[(size_t)e * H + h] = lg;
}

// exp(logit - max) + segment sum
__global__ void k_expden(const float* __restrict__ logit, const unsigned* __restrict__ mxkey,
                         const int* __restrict__ src, const int* __restrict__ dst,
                         const int* ecount, int Emax, int n, int H,
                         float* __restrict__ ex, float* __restrict__ den) {
  long long t = (long long)blockIdx.x * blockDim.x + threadIdx.x;
  long long total = (long long)(Emax + n) * H;
  if (t >= total) return;
  int h = (int)(t % H);
  long long e = t / H;
  int s, d;
  float v = 0.f;
  if (edge_decode(e, src, dst, ecount, Emax, s, d)) {
    float mx = key2f(mxkey[(size_t)d * H + h]);
    v = __expf(logit[(size_t)e * H + h] - mx);
    atomicAdd(&den[(size_t)d * H + h], v);
  }
  ex[(size_t)e * H + h] = v;
}

// scatter: out[d] += xh[s] * alpha   (one block = one edge, 128 feature lanes)
__global__ __launch_bounds__(128) void k_scatter(const float* __restrict__ xh,
                                                 const float* __restrict__ ex,
                                                 const float* __restrict__ den,
                                                 const int* __restrict__ src,
                                                 const int* __restrict__ dst,
                                                 const int* ecount, int Emax, int n, int H,
                                                 float* __restrict__ out) {
  long long e = blockIdx.x;
  int f = threadIdx.x;
  int s, d;
  if (!edge_decode(e, src, dst, ecount, Emax, s, d)) return;
  for (int h = 0; h < H; ++h) {
    float exv = ex[(size_t)e * H + h];
    if (exv != 0.f) {
      float alpha = exv / den[(size_t)d * H + h];
      atomicAdd(&out[((size_t)d * H + h) * DFEAT + f],
                xh[((size_t)s * H + h) * DFEAT + f] * alpha);
    }
  }
}

// out = tanh(out + bias), float4 vectorized (n*F multiple of 4, F multiple of 4)
__global__ void k_bias_tanh(float* __restrict__ out, const float* __restrict__ bias,
                            long long n, int F) {
  long long i4 = ((long long)blockIdx.x * blockDim.x + threadIdx.x) * 4;
  if (i4 >= n * F) return;
  int f = (int)(i4 % F);
  v4f v = *(const v4f*)&out[i4];
  v4f b = *(const v4f*)&bias[f];
#pragma unroll
  for (int j = 0; j < 4; ++j) v[j] = tanhf(v[j] + b[j]);
  *(v4f*)&out[i4] = v;
}

// BatchNorm stats: blockDim == F, each block reduces a 512-row chunk
__global__ void k_bn_stats(const float* __restrict__ h, float* __restrict__ sums,
                           float* __restrict__ sqs, int n, int F) {
  int f = threadIdx.x;
  int r0 = blockIdx.x * 512;
  int r1 = r0 + 512; if (r1 > n) r1 = n;
  float s = 0.f, q = 0.f;
  for (int r = r0; r < r1; ++r) {
    float v = h[(size_t)r * F + f];
    s += v; q += v * v;
  }
  atomicAdd(&sums[f], s);
  atomicAdd(&sqs[f], q);
}

__global__ void k_bn_apply(float* __restrict__ h, const float* __restrict__ sums,
                           const float* __restrict__ sqs, const float* __restrict__ g,
                           const float* __restrict__ b, int n, int F) {
  long long i4 = ((long long)blockIdx.x * blockDim.x + threadIdx.x) * 4;
  if (i4 >= (long long)n * F) return;
  int f = (int)(i4 % F);
  float inv_n = 1.0f / (float)n;
  v4f v = *(const v4f*)&h[i4];
#pragma unroll
  for (int j = 0; j < 4; ++j) {
    float mean = sums[f + j] * inv_n;
    float var  = sqs[f + j] * inv_n - mean * mean;
    v[j] = (v[j] - mean) * rsqrtf(var + 1e-5f) * g[f + j] + b[f + j];
  }
  *(v4f*)&h[i4] = v;
}

// 1/||p||
__global__ __launch_bounds__(256) void k_rnorm(const float* __restrict__ p,
                                               float* __restrict__ invn, int F) {
  __shared__ float sh[256];
  int t = threadIdx.x;
  float s = 0.f;
  for (int i = t; i < F; i += 256) { float v = p[i]; s += v * v; }
  sh[t] = s; __syncthreads();
  for (int o = 128; o; o >>= 1) { if (t < o) sh[t] += sh[t + o]; __syncthreads(); }
  if (t == 0) invn[0] = rsqrtf(sh[0]);
}

// score = tanh(h . p / ||p||); one wave per node
__global__ __launch_bounds__(256) void k_score(const float* __restrict__ h,
                                               const float* __restrict__ p,
                                               const float* __restrict__ invn,
                                               float* __restrict__ score,
                                               unsigned* __restrict__ keys,
                                               int n, int F) {
  long long gid = (long long)blockIdx.x * blockDim.x + threadIdx.x;
  long long wv  = gid >> 5;
  int lane = threadIdx.x & 31;
  if (wv >= n) return;
  const float* row = h + (size_t)wv * F;
  float s = 0.f;
  for (int c = lane; c < F; c += 32) s += row[c] * p[c];
  for (int off = 16; off; off >>= 1) s += __shfl_down(s, off);
  if (lane == 0) {
    float sc = tanhf(s * invn[0]);
    score[wv] = sc;
    keys[wv]  = f2key(sc);
  }
}

// ------------------------- radix threshold select --------------------------
struct SelState { unsigned prefix; int remain; };

__global__ void k_sel_init(SelState* st, int K) {
  if (threadIdx.x == 0 && blockIdx.x == 0) { st->prefix = 0u; st->remain = K; }
}

__global__ __launch_bounds__(256) void k_hist(const unsigned* __restrict__ keys, int n,
                                              const SelState* __restrict__ st,
                                              unsigned mask, int shift,
                                              unsigned* __restrict__ hist) {
  __shared__ unsigned sh[256];
  int t = threadIdx.x;
  sh[t] = 0;
  __syncthreads();
  unsigned pre = st->prefix;
  for (long long i = (long long)blockIdx.x * blockDim.x + t; i < n;
       i += (long long)gridDim.x * blockDim.x) {
    unsigned k = keys[i];
    if ((k & mask) == pre) atomicAdd(&sh[(k >> shift) & 255u], 1u);
  }
  __syncthreads();
  if (sh[t]) atomicAdd(&hist[t], sh[t]);
}

__global__ void k_pick(const unsigned* __restrict__ hist, SelState* st, int shift) {
  if (threadIdx.x != 0 || blockIdx.x != 0) return;
  int remain = st->remain;
  for (int b = 255; b >= 0; --b) {
    int c = (int)hist[b];
    if (c >= remain) {
      st->prefix |= ((unsigned)b) << shift;
      st->remain = remain;
      return;
    }
    remain -= c;
  }
}

// compact the top-K set: node_map[orig] = new index (or -1), perm[new] = orig
__global__ void k_compact(const unsigned* __restrict__ keys, int n,
                          const SelState* __restrict__ st, int K,
                          int* __restrict__ node_map, int* __restrict__ perm,
                          int* __restrict__ counters) {
  long long i = (long long)blockIdx.x * blockDim.x + threadIdx.x;
  if (i >= n) return;
  unsigned k = keys[i];
  unsigned T = st->prefix;
  int remain = st->remain;
  int pos = -1;
  if (k > T) {
    pos = atomicAdd(&counters[0], 1);
  } else if (k == T) {
    int t = atomicAdd(&counters[1], 1);
    if (t < remain) pos = (K - remain) + t;
  }
  node_map[i] = pos;
  if (pos >= 0) perm[pos] = (int)i;
}

// gather + gate (+ optional relu), float4: outp[new] = relu?(h[perm]*score[perm])
__global__ void k_gather(const float* __restrict__ h, const float* __restrict__ score,
                         const int* __restrict__ perm, float* __restrict__ outp,
                         int Kn, int F, int do_relu) {
  long long i4 = ((long long)blockIdx.x * blockDim.x + threadIdx.x) * 4;
  if (i4 >= (long long)Kn * F) return;
  int row = (int)(i4 / F), f = (int)(i4 % F);
  int orig = perm[row];
  float sc = score[orig];
  v4f v = *(const v4f*)&h[(size_t)orig * F + f];
#pragma unroll
  for (int j = 0; j < 4; ++j) {
    float w = v[j] * sc;
    if (do_relu && w < 0.f) w = 0.f;
    v[j] = w;
  }
  *(v4f*)&outp[i4] = v;
}

// relabel edges of pooled graph (drops self loops + dead endpoints, compacts)
__global__ void k_relabel(const int* __restrict__ src, const int* __restrict__ dst,
                          int E, const int* __restrict__ node_map,
                          int* __restrict__ osrc, int* __restrict__ odst,
                          int* __restrict__ ecnt) {
  long long e = (long long)blockIdx.x * blockDim.x + threadIdx.x;
  if (e >= E) return;
  int s = src[e], d = dst[e];
  if (s == d) return;
  int ns = node_map[s], nd = node_map[d];
  if (ns < 0 || nd < 0) return;
  int pos = atomicAdd(ecnt, 1);
  osrc[pos] = ns;
  odst[pos] = nd;
}

// global max pool over selected nodes of (h * score); keys in monotonic uint
__global__ __launch_bounds__(128) void k_maxpool(const float* __restrict__ h,
                                                 const float* __restrict__ score,
                                                 const unsigned* __restrict__ keys,
                                                 const SelState* __restrict__ st,
                                                 int n, int* __restrict__ tiecnt,
                                                 unsigned* __restrict__ gmax) {
  int i = blockIdx.x;
  if (i >= n) return;
  __shared__ int take;
  if (threadIdx.x == 0) {
    unsigned k = keys[i], T = st->prefix;
    int tk = 0;
    if (k > T) tk = 1;
    else if (k == T) { int t = atomicAdd(tiecnt, 1); if (t < st->remain) tk = 1; }
    take = tk;
  }
  __syncthreads();
  if (!take) return;
  float v = h[(size_t)i * DFEAT + threadIdx.x] * score[i];
  atomicMax(&gmax[threadIdx.x], f2key(v));
}

// final MLP: relu(max) -> 256 (relu) -> 128
__global__ __launch_bounds__(256) void k_mlp(const unsigned* __restrict__ gmax,
                                             const float* __restrict__ Wm1,
                                             const float* __restrict__ bm1,
                                             const float* __restrict__ Wm2,
                                             const float* __restrict__ bm2,
                                             float* __restrict__ out) {
  __shared__ float v[128];
  __shared__ float hid[256];
  int t = threadIdx.x;
  if (t < 128) { float x = key2f(gmax[t]); v[t] = (x > 0.f) ? x : 0.f; }
  __syncthreads();
  float a = bm1[t];
  for (int i = 0; i < 128; ++i) a += v[i] * Wm1[i * 256 + t];
  hid[t] = (a > 0.f) ? a : 0.f;
  __syncthreads();
  if (t < 128) {
    float o = bm2[t];
    for (int i = 0; i < 256; ++i) o += hid[i] * Wm2[i * 128 + t];
    out[t] = o;
  }
}

// ---------------------------------------------------------------------------
// Host orchestration
// ---------------------------------------------------------------------------
static inline int cdiv64(long long a, long long b) { return (int)((a + b - 1) / b); }

extern "C" void kernel_launch(void* const* d_in, const int* in_sizes, int n_in,
                              void* d_out, int out_size, void* d_ws, size_t ws_size,
                              hipStream_t stream) {
  (void)in_sizes; (void)n_in; (void)out_size; (void)ws_size;
  const int N = NNODES, E = NEDGES;

  const float* x   = (const float*)d_in[0];
  const int*   ei  = (const int*)d_in[1];
  const float* W0  = (const float*)d_in[3];
  const float* as0 = (const float*)d_in[4];
  const float* ad0 = (const float*)d_in[5];
  const float* b0  = (const float*)d_in[6];
  const float* g0  = (const float*)d_in[7];
  const float* be0 = (const float*)d_in[8];
  const float* p0  = (const float*)d_in[9];
  const float* W1  = (const float*)d_in[10];
  const float* as1 = (const float*)d_in[11];
  const float* ad1 = (const float*)d_in[12];
  const float* b1  = (const float*)d_in[13];
  const float* g1  = (const float*)d_in[14];
  const float* be1 = (const float*)d_in[15];
  const float* p1  = (const float*)d_in[16];
  const float* Wm1 = (const float*)d_in[17];
  const float* bm1 = (const float*)d_in[18];
  const float* Wm2 = (const float*)d_in[19];
  const float* bm2 = (const float*)d_in[20];
  const int* src = ei;
  const int* dst = ei + E;

  // ---- workspace bump allocator ----
  char* wp = (char*)d_ws;
  auto alloc = [&](size_t bytes) -> void* {
    void* r = (void*)wp;
    wp += (bytes + 255) & ~(size_t)255;
    return r;
  };
  float*    xh0    = (float*)alloc((size_t)N * 384 * 4);   // reused as h1in
  float*    h0     = (float*)alloc((size_t)N * 384 * 4);
  float*    xh1    = (float*)alloc((size_t)K0SEL * 128 * 4);
  float*    h1     = (float*)alloc((size_t)K0SEL * 128 * 4);
  float*    als0   = (float*)alloc((size_t)N * 3 * 4);
  float*    ald0   = (float*)alloc((size_t)N * 3 * 4);
  unsigned* mx0    = (unsigned*)alloc((size_t)N * 3 * 4);
  float*    den0   = (float*)alloc((size_t)N * 3 * 4);
  float*    logit0 = (float*)alloc((size_t)(E + N) * 3 * 4);
  float*    ex0    = (float*)alloc((size_t)(E + N) * 3 * 4);
  float*    score0 = (float*)alloc((size_t)N * 4);
  unsigned* keys0  = (unsigned*)alloc((size_t)N * 4);
  int*      nmap0  = (int*)alloc((size_t)N * 4);
  int*      perm0  = (int*)alloc((size_t)K0SEL * 4);
  int*      esrc1  = (int*)alloc((size_t)E * 4);
  int*      edst1  = (int*)alloc((size_t)E * 4);
  float*    als1   = (float*)alloc((size_t)K0SEL * 4);
  float*    ald1   = (float*)alloc((size_t)K0SEL * 4);
  unsigned* mx1    = (unsigned*)alloc((size_t)K0SEL * 4);
  float*    den1   = (float*)alloc((size_t)K0SEL * 4);
  float*    logit1 = (float*)alloc((size_t)(E + K0SEL) * 4);
  float*    ex1    = (float*)alloc((size_t)(E + K0SEL) * 4);
  float*    score1 = (float*)alloc((size_t)K0SEL * 4);
  unsigned* keys1  = (unsigned*)alloc((size_t)K0SEL * 4);
  unsigned* hist   = (unsigned*)alloc(256 * 4);
  SelState* st     = (SelState*)alloc(sizeof(SelState));
  int*      cnt2   = (int*)alloc(2 * 4);
  int*      ecnt   = (int*)alloc(4);
  int*      tie    = (int*)alloc(4);
  float*    invn   = (float*)alloc(4);
  float*    bnsum  = (float*)alloc(384 * 4);
  float*    bnsq   = (float*)alloc(384 * 4);
  unsigned* gmax   = (unsigned*)alloc(128 * 4);

  const unsigned selMask[4]  = {0x00000000u, 0xFF000000u, 0xFFFF0000u, 0xFFFFFF00u};
  const int      selShift[4] = {24, 16, 8, 0};

  // ===================== Layer 0: GAT (heads=3) =====================
  k_gemm_wmma<<<dim3(N / TM, 384 / TN), 256, 0, stream>>>(x, W0, xh0, N, 128, 384);
  k_att<<<cdiv64((long long)N * 3 * 32, 256), 256, 0, stream>>>(xh0, as0, ad0, als0, ald0, N, 3);

  hipMemsetAsync(mx0,  0, (size_t)N * 3 * 4, stream);
  hipMemsetAsync(den0, 0, (size_t)N * 3 * 4, stream);
  hipMemsetAsync(h0,   0, (size_t)N * 384 * 4, stream);

  k_logits<<<cdiv64((long long)(E + N) * 3, 256), 256, 0, stream>>>(
      als0, ald0, src, dst, nullptr, E, N, 3, logit0, mx0);
  k_expden<<<cdiv64((long long)(E + N) * 3, 256), 256, 0, stream>>>(
      logit0, mx0, src, dst, nullptr, E, N, 3, ex0, den0);
  k_scatter<<<E + N, 128, 0, stream>>>(xh0, ex0, den0, src, dst, nullptr, E, N, 3, h0);
  k_bias_tanh<<<cdiv64((long long)N * 384 / 4, 256), 256, 0, stream>>>(h0, b0, N, 384);

  // BatchNorm 0
  hipMemsetAsync(bnsum, 0, 384 * 4, stream);
  hipMemsetAsync(bnsq,  0, 384 * 4, stream);
  k_bn_stats<<<cdiv64(N, 512), 384, 0, stream>>>(h0, bnsum, bnsq, N, 384);
  k_bn_apply<<<cdiv64((long long)N * 384 / 4, 256), 256, 0, stream>>>(h0, bnsum, bnsq, g0, be0, N, 384);

  // TopK pool 0
  k_rnorm<<<1, 256, 0, stream>>>(p0, invn, 384);
  k_score<<<cdiv64((long long)N * 32, 256), 256, 0, stream>>>(h0, p0, invn, score0, keys0, N, 384);
  k_sel_init<<<1, 1, 0, stream>>>(st, K0SEL);
  for (int r = 0; r < 4; ++r) {
    hipMemsetAsync(hist, 0, 256 * 4, stream);
    k_hist<<<256, 256, 0, stream>>>(keys0, N, st, selMask[r], selShift[r], hist);
    k_pick<<<1, 1, 0, stream>>>(hist, st, selShift[r]);
  }
  hipMemsetAsync(cnt2, 0, 2 * 4, stream);
  k_compact<<<cdiv64(N, 256), 256, 0, stream>>>(keys0, N, st, K0SEL, nmap0, perm0, cnt2);

  float* h1in = xh0;  // xh0 is dead now; reuse as pooled layer-1 input
  k_gather<<<cdiv64((long long)K0SEL * 384 / 4, 256), 256, 0, stream>>>(
      h0, score0, perm0, h1in, K0SEL, 384, /*relu=*/1);
  hipMemsetAsync(ecnt, 0, 4, stream);
  k_relabel<<<cdiv64(E, 256), 256, 0, stream>>>(src, dst, E, nmap0, esrc1, edst1, ecnt);

  // ===================== Layer 1: GAT (heads=1) =====================
  k_gemm_wmma<<<dim3(cdiv64(K0SEL, TM), 128 / TN), 256, 0, stream>>>(h1in, W1, xh1, K0SEL, 384, 128);
  k_att<<<cdiv64((long long)K0SEL * 32, 256), 256, 0, stream>>>(xh1, as1, ad1, als1, ald1, K0SEL, 1);

  hipMemsetAsync(mx1,  0, (size_t)K0SEL * 4, stream);
  hipMemsetAsync(den1, 0, (size_t)K0SEL * 4, stream);
  hipMemsetAsync(h1,   0, (size_t)K0SEL * 128 * 4, stream);

  k_logits<<<cdiv64((long long)(E + K0SEL), 256), 256, 0, stream>>>(
      als1, ald1, esrc1, edst1, ecnt, E, K0SEL, 1, logit1, mx1);
  k_expden<<<cdiv64((long long)(E + K0SEL), 256), 256, 0, stream>>>(
      logit1, mx1, esrc1, edst1, ecnt, E, K0SEL, 1, ex1, den1);
  k_scatter<<<E + K0SEL, 128, 0, stream>>>(xh1, ex1, den1, esrc1, edst1, ecnt, E, K0SEL, 1, h1);
  k_bias_tanh<<<cdiv64((long long)K0SEL * 128 / 4, 256), 256, 0, stream>>>(h1, b1, K0SEL, 128);

  // BatchNorm 1
  hipMemsetAsync(bnsum, 0, 128 * 4, stream);
  hipMemsetAsync(bnsq,  0, 128 * 4, stream);
  k_bn_stats<<<cdiv64(K0SEL, 512), 128, 0, stream>>>(h1, bnsum, bnsq, K0SEL, 128);
  k_bn_apply<<<cdiv64((long long)K0SEL * 128 / 4, 256), 256, 0, stream>>>(h1, bnsum, bnsq, g1, be1, K0SEL, 128);

  // TopK pool 1 (selection only; order irrelevant for global max pool)
  k_rnorm<<<1, 256, 0, stream>>>(p1, invn, 128);
  k_score<<<cdiv64((long long)K0SEL * 32, 256), 256, 0, stream>>>(h1, p1, invn, score1, keys1, K0SEL, 128);
  k_sel_init<<<1, 1, 0, stream>>>(st, K1SEL);
  for (int r = 0; r < 4; ++r) {
    hipMemsetAsync(hist, 0, 256 * 4, stream);
    k_hist<<<256, 256, 0, stream>>>(keys1, K0SEL, st, selMask[r], selShift[r], hist);
    k_pick<<<1, 1, 0, stream>>>(hist, st, selShift[r]);
  }

  // global max pool over selected nodes of (h * score), then relu + MLP
  hipMemsetAsync(tie,  0, 4, stream);
  hipMemsetAsync(gmax, 0, 128 * 4, stream);
  k_maxpool<<<K0SEL, 128, 0, stream>>>(h1, score1, keys1, st, K0SEL, tie, gmax);
  k_mlp<<<1, 256, 0, stream>>>(gmax, Wm1, bm1, Wm2, bm2, (float*)d_out);
}
